// AutoDisBucketEncoder_69535520522832
// MI455X (gfx1250) — compile-verified, matrix operand order checked by reading, not run on hardware
//
#include <hip/hip_runtime.h>
#include <hip/hip_bf16.h>

#define B_TOT 2048
#define F_TOT 256
#define D_DIM 64
#define E_DIM 128
#define NEG_SLOPE 0.01f

typedef __attribute__((ext_vector_type(16))) _Float16 v16h;
typedef __attribute__((ext_vector_type(8)))  _Float16 v8h;
typedef __attribute__((ext_vector_type(8)))  float    v8f;

union V16 { v16h v; v8h h8[2]; unsigned u[8]; };

struct __align__(16) SMem {
  unsigned rwswz[6144];     // 3 layers * 4 ntiles * 2 kchunks * 32 lanes * 8 dwords
  unsigned w2swz[512];      // 1 ntile (N padded 8->16) * 2 kchunks
  unsigned embswz[2048];    // 8 ntiles * 1 kchunk (K padded 8->32)
  float w1s[192];
  float b1s[64];
  float rbs[3][64];
  float b2s[8];
  float taus[8];
  _Float16 hst[8][1024];    // per-wave 16x64 f16 restage buffer
  float xs[8][16][3];
  float lg[8][128];         // per-wave 16x8 logits
};

__device__ __forceinline__ float leaky(float v) { return fmaxf(v, NEG_SLOPE * v); }

__device__ __forceinline__ unsigned pk2(float a, float b) {
  union { unsigned u; _Float16 h[2]; } x;
  x.h[0] = (_Float16)a; x.h[1] = (_Float16)b;
  return x.u;
}

// A fragment (16x32 f16) from row-major 16x64 f16 LDS tile: two ds_load_b128 per lane.
__device__ __forceinline__ v16h load_afrag(const _Float16* hrow, int lane, int chunk) {
  const int m = lane & 15, hiL = lane >> 4;
  const _Float16* base = hrow + m * 64 + hiL * 8 + chunk * 32;
  V16 r;
  r.h8[0] = *(const v8h*)base;          // K = chunk*32 + (hi?8:0) .. +7
  r.h8[1] = *(const v8h*)(base + 16);   // K = chunk*32 + 16 + (hi?8:0) .. +7
  return r.v;
}

// B fragment (32x16 f16) from pre-swizzled LDS: contiguous 32B per lane.
__device__ __forceinline__ v16h load_bfrag(const unsigned* swz, int fragIdx, int lane) {
  const unsigned* p = swz + fragIdx * 256 + lane * 8;
  V16 r;
  r.h8[0] = *(const v8h*)p;
  r.h8[1] = *(const v8h*)(p + 4);
  return r.v;
}

__global__ __launch_bounds__(256)
void autodis_kernel(const float* __restrict__ x,  const float* __restrict__ w1,
                    const float* __restrict__ b1, const float* __restrict__ rw0,
                    const float* __restrict__ rb0,const float* __restrict__ rw1,
                    const float* __restrict__ rb1,const float* __restrict__ rw2,
                    const float* __restrict__ rb2,const float* __restrict__ w2,
                    const float* __restrict__ b2, const float* __restrict__ tau,
                    const float* __restrict__ emb,float* __restrict__ out) {
  __shared__ SMem sm;
  const int tid  = threadIdx.x;
  const int f    = blockIdx.y;
  const int lane = tid & 31, wave = tid >> 5;
  const int ncol = lane & 15, hiL = lane >> 4;

  const float* rw0b = rw0 + f * 4096;
  const float* rw1b = rw1 + f * 4096;
  const float* rw2b = rw2 + f * 4096;
  const float* w2b  = w2  + f * 512;
  const float* embb = emb + f * 1024;

  // ---- cooperative weight staging (f32 -> f16, B-fragment swizzle) ----
  for (int idx = tid; idx < 6144; idx += 256) {
    const int slot = idx & 7, ln = (idx >> 3) & 31, kc = (idx >> 8) & 1;
    const int nt = (idx >> 9) & 3, layer = idx >> 11;
    const float* rwp = (layer == 0) ? rw0b : (layer == 1) ? rw1b : rw2b;
    const int n = nt * 16 + (ln & 15);
    const int k = kc * 32 + 2 * (slot & 3) + ((ln & 16) ? 8 : 0) + ((slot & 4) ? 16 : 0);
    sm.rwswz[idx] = pk2(rwp[k * 64 + n], rwp[(k + 1) * 64 + n]);
  }
  for (int idx = tid; idx < 512; idx += 256) {
    const int slot = idx & 7, ln = (idx >> 3) & 31, kc = (idx >> 8) & 1;
    const int n = ln & 15;
    const int k = kc * 32 + 2 * (slot & 3) + ((ln & 16) ? 8 : 0) + ((slot & 4) ? 16 : 0);
    const float a = (n < 8) ? w2b[k * 8 + n] : 0.f;
    const float b = (n < 8) ? w2b[(k + 1) * 8 + n] : 0.f;
    sm.w2swz[idx] = pk2(a, b);
  }
  for (int idx = tid; idx < 2048; idx += 256) {
    const int slot = idx & 7, ln = (idx >> 3) & 31, nt = (idx >> 8) & 7;
    const int n = nt * 16 + (ln & 15);
    const int k = 2 * (slot & 3) + ((ln & 16) ? 8 : 0) + ((slot & 4) ? 16 : 0);
    const float a = (k < 8)     ? embb[k * 128 + n]       : 0.f;
    const float b = (k + 1 < 8) ? embb[(k + 1) * 128 + n] : 0.f;
    sm.embswz[idx] = pk2(a, b);
  }
  for (int i = tid; i < 192; i += 256) sm.w1s[i] = w1[f * 192 + i];
  for (int i = tid; i < 64; i += 256) {
    sm.b1s[i]    = b1[f * 64 + i];
    sm.rbs[0][i] = rb0[f * 64 + i];
    sm.rbs[1][i] = rb1[f * 64 + i];
    sm.rbs[2][i] = rb2[f * 64 + i];
  }
  if (tid < 8) { sm.b2s[tid] = b2[f * 8 + tid]; sm.taus[tid] = tau[f * 8 + tid]; }
  __syncthreads();

  // ---- per-wave feature-level constants ----
  float w1c[4][3], b1c[4], rbc[3][4];
  #pragma unroll
  for (int nt = 0; nt < 4; ++nt) {
    #pragma unroll
    for (int i = 0; i < 3; ++i) w1c[nt][i] = sm.w1s[i * 64 + nt * 16 + ncol];
    b1c[nt] = sm.b1s[nt * 16 + ncol];
  }
  #pragma unroll
  for (int l = 0; l < 3; ++l)
    #pragma unroll
    for (int nt = 0; nt < 4; ++nt) rbc[l][nt] = sm.rbs[l][nt * 16 + ncol];
  const float b2c  = sm.b2s[ncol & 7];
  const float tauc = sm.taus[ncol & 7];
  _Float16* hrow = sm.hst[wave];

  for (int t = 0; t < 8; ++t) {
    const int tile = (blockIdx.x * 8 + wave) * 8 + t;
    const int b0 = tile * 16;

    // stage 16 rows of x (3 f32 each) into LDS
    if (lane < 16) {
      const float* xp = x + ((size_t)(b0 + lane) * F_TOT + f) * 3;
      sm.xs[wave][lane][0] = xp[0];
      sm.xs[wave][lane][1] = xp[1];
      sm.xs[wave][lane][2] = xp[2];
      if (t + 1 < 8)
        __builtin_prefetch(x + ((size_t)(b0 + 16 + lane) * F_TOT + f) * 3, 0, 1);
    }

    // layer 1: K=3, VALU. hacc in WMMA C layout (4 ntiles x v8f).
    v8f hacc[4];
    #pragma unroll
    for (int r = 0; r < 8; ++r) {
      const int m = r + 8 * hiL;
      const float x0 = sm.xs[wave][m][0];
      const float x1 = sm.xs[wave][m][1];
      const float x2 = sm.xs[wave][m][2];
      #pragma unroll
      for (int nt = 0; nt < 4; ++nt)
        hacc[nt][r] = leaky(fmaf(x0, w1c[nt][0],
                            fmaf(x1, w1c[nt][1],
                            fmaf(x2, w1c[nt][2], b1c[nt]))));
    }

    // 3 residual layers: h = leaky(h@rw + rb + h)
    #pragma unroll
    for (int layer = 0; layer < 3; ++layer) {
      #pragma unroll
      for (int nt = 0; nt < 4; ++nt)
        #pragma unroll
        for (int r = 0; r < 8; ++r)
          hrow[(r + 8 * hiL) * 64 + nt * 16 + ncol] = (_Float16)hacc[nt][r];
      const v16h a0 = load_afrag(hrow, lane, 0);
      const v16h a1 = load_afrag(hrow, lane, 1);
      #pragma unroll
      for (int nt = 0; nt < 4; ++nt) {
        const v16h bf0 = load_bfrag(sm.rwswz, (layer * 4 + nt) * 2 + 0, lane);
        const v16h bf1 = load_bfrag(sm.rwswz, (layer * 4 + nt) * 2 + 1, lane);
        v8f c;
        #pragma unroll
        for (int r = 0; r < 8; ++r) c[r] = hacc[nt][r] + rbc[layer][nt];  // residual + bias
        c = __builtin_amdgcn_wmma_f32_16x16x32_f16(false, a0, false, bf0, (short)0, c, false, false);
        c = __builtin_amdgcn_wmma_f32_16x16x32_f16(false, a1, false, bf1, (short)0, c, false, false);
        #pragma unroll
        for (int r = 0; r < 8; ++r) hacc[nt][r] = leaky(c[r]);
      }
    }

    // logits = leaky(h @ w2 + b2) * tau  (N padded to 16)
    #pragma unroll
    for (int nt = 0; nt < 4; ++nt)
      #pragma unroll
      for (int r = 0; r < 8; ++r)
        hrow[(r + 8 * hiL) * 64 + nt * 16 + ncol] = (_Float16)hacc[nt][r];
    const v16h a0 = load_afrag(hrow, lane, 0);
    const v16h a1 = load_afrag(hrow, lane, 1);
    const v16h wf0 = load_bfrag(sm.w2swz, 0, lane);
    const v16h wf1 = load_bfrag(sm.w2swz, 1, lane);
    v8f lc;
    #pragma unroll
    for (int r = 0; r < 8; ++r) lc[r] = 0.f;
    lc = __builtin_amdgcn_wmma_f32_16x16x32_f16(false, a0, false, wf0, (short)0, lc, false, false);
    lc = __builtin_amdgcn_wmma_f32_16x16x32_f16(false, a1, false, wf1, (short)0, lc, false, false);
    if (ncol < 8) {
      #pragma unroll
      for (int r = 0; r < 8; ++r)
        sm.lg[wave][(r + 8 * hiL) * 8 + ncol] = leaky(lc[r] + b2c) * tauc;
    }

    // softmax over K=8: lanes 0-15 each own one row; pack p into zero-padded A frag
    unsigned pku[4] = {0u, 0u, 0u, 0u};
    if (lane < 16) {
      float vv[8];
      float mx = -3.4e38f;
      #pragma unroll
      for (int k = 0; k < 8; ++k) { vv[k] = sm.lg[wave][lane * 8 + k]; mx = fmaxf(mx, vv[k]); }
      float s = 0.f;
      #pragma unroll
      for (int k = 0; k < 8; ++k) { vv[k] = __expf(vv[k] - mx); s += vv[k]; }
      const float inv = 1.0f / s;
      #pragma unroll
      for (int j = 0; j < 4; ++j) pku[j] = pk2(vv[2 * j] * inv, vv[2 * j + 1] * inv);
    }
    V16 pa;
    #pragma unroll
    for (int s2 = 0; s2 < 8; ++s2) pa.u[s2] = (s2 < 4) ? pku[s2] : 0u;

    // out tile = p @ emb  (K padded 8->32), store f32 straight from accumulators
    float* op = out + (size_t)b0 * (F_TOT * E_DIM) + (size_t)f * E_DIM;
    #pragma unroll
    for (int nt = 0; nt < 8; ++nt) {
      const v16h ef = load_bfrag(sm.embswz, nt, lane);
      v8f oc;
      #pragma unroll
      for (int r = 0; r < 8; ++r) oc[r] = 0.f;
      oc = __builtin_amdgcn_wmma_f32_16x16x32_f16(false, pa.v, false, ef, (short)0, oc, false, false);
      #pragma unroll
      for (int r = 0; r < 8; ++r)
        op[(size_t)(r + 8 * hiL) * (F_TOT * E_DIM) + nt * 16 + ncol] = oc[r];
    }
  }
}

extern "C" void kernel_launch(void* const* d_in, const int* in_sizes, int n_in,
                              void* d_out, int out_size, void* d_ws, size_t ws_size,
                              hipStream_t stream) {
  const float* x   = (const float*)d_in[0];
  const float* w1  = (const float*)d_in[1];
  const float* b1  = (const float*)d_in[2];
  const float* rw0 = (const float*)d_in[3];
  const float* rb0 = (const float*)d_in[4];
  const float* rw1 = (const float*)d_in[5];
  const float* rb1 = (const float*)d_in[6];
  const float* rw2 = (const float*)d_in[7];
  const float* rb2 = (const float*)d_in[8];
  const float* w2  = (const float*)d_in[9];
  const float* b2  = (const float*)d_in[10];
  const float* tau = (const float*)d_in[11];
  const float* emb = (const float*)d_in[12];
  float* out = (float*)d_out;

  dim3 grid(2, F_TOT, 1);   // 2 batch super-tiles x 256 features
  dim3 block(256, 1, 1);    // 8 waves; each wave: 8 tiles of 16 rows
  autodis_kernel<<<grid, block, 0, stream>>>(x, w1, b1, rw0, rb0, rw1, rb1,
                                             rw2, rb2, w2, b2, tau, emb, out);
}
